// MemoryModel_35270271435165
// MI455X (gfx1250) — matrix operand order repaired and checked by reading, not compiled, not run on hardware
//
#include <hip/hip_runtime.h>
#include <hip/hip_bf16.h>

// ---------------- types ----------------
typedef __attribute__((ext_vector_type(16))) __bf16 v16bf;
typedef __attribute__((ext_vector_type(8)))  __bf16 v8bf;
typedef __attribute__((ext_vector_type(8)))  float  v8f;

#define B_SZ    32
#define L_SZ    4096
#define HID     512
#define VOCAB_N 32000
#define SLOTS   256
#define KSEL    128
#define ROWS    (B_SZ * L_SZ)   // 131072

// ---------------- weight transpose + bf16 convert: W[K][N] -> Wt[N][K] ----------------
__global__ __launch_bounds__(256) void transpose_bf16_kernel(
    const float* __restrict__ W, __bf16* __restrict__ Wt, int K, int N)
{
    size_t i = (size_t)blockIdx.x * blockDim.x + threadIdx.x;
    size_t total = (size_t)K * N;
    if (i >= total) return;
    int k = (int)(i / N);
    int n = (int)(i % N);
    Wt[(size_t)n * K + k] = (__bf16)W[i];
}

// ---------------- plain fp32 -> bf16 convert (same layout) ----------------
__global__ __launch_bounds__(256) void convert_bf16_kernel(
    const float* __restrict__ W, __bf16* __restrict__ Wb, size_t total)
{
    size_t i = (size_t)blockIdx.x * blockDim.x + threadIdx.x;
    if (i < total) Wb[i] = (__bf16)W[i];
}

// ---------------- fragment loaders ----------------
// ISA 16-bit A layout (16x32): lanes 0-15 hold K k..k+7 & k+16..k+23,
// lanes 16-31 hold K k+8..k+15 & k+24..k+31.
__device__ __forceinline__ v16bf load_a_f32(const float* __restrict__ Arow, int k, int hi)
{
    const int klo = k + hi * 8;
    const float4 a0 = *(const float4*)(Arow + klo);
    const float4 a1 = *(const float4*)(Arow + klo + 4);
    const float4 a2 = *(const float4*)(Arow + klo + 16);
    const float4 a3 = *(const float4*)(Arow + klo + 20);
    v16bf af;
    af[0]  = (__bf16)a0.x; af[1]  = (__bf16)a0.y; af[2]  = (__bf16)a0.z; af[3]  = (__bf16)a0.w;
    af[4]  = (__bf16)a1.x; af[5]  = (__bf16)a1.y; af[6]  = (__bf16)a1.z; af[7]  = (__bf16)a1.w;
    af[8]  = (__bf16)a2.x; af[9]  = (__bf16)a2.y; af[10] = (__bf16)a2.z; af[11] = (__bf16)a2.w;
    af[12] = (__bf16)a3.x; af[13] = (__bf16)a3.y; af[14] = (__bf16)a3.z; af[15] = (__bf16)a3.w;
    return af;
}

__device__ __forceinline__ v16bf load_a_bf16(const __bf16* __restrict__ Arow, int k, int hi)
{
    const int klo = k + hi * 8;
    const v8bf lo = *(const v8bf*)(Arow + klo);
    const v8bf hc = *(const v8bf*)(Arow + klo + 16);
    v16bf af;
#pragma unroll
    for (int i = 0; i < 8; ++i) { af[i] = lo[i]; af[8 + i] = hc[i]; }
    return af;
}

// B fragment (32x16 bf16): lane = column; lanes 0-15 hold K k..k+7 & k+16..k+23,
// lanes 16-31 hold K k+8..k+15 & k+24..k+31; Wt is [N][K] (two 16 B runs).
__device__ __forceinline__ v16bf load_b(const __bf16* __restrict__ Wt, size_t col, int K, int k, int hi)
{
    const __bf16* p = Wt + col * (size_t)K + (k + hi * 8);
    const v8bf lo = *(const v8bf*)(p);
    const v8bf hc = *(const v8bf*)(p + 16);
    v16bf bf;
#pragma unroll
    for (int i = 0; i < 8; ++i) { bf[i] = lo[i]; bf[8 + i] = hc[i]; }
    return bf;
}

// ---------------- WMMA GEMM: out[M,N] = act(A[rowIdx[m] or m, K] @ Wt^T + bias) ----------------
// WG = 256 threads = 8 waves; WG tile 32(M) x 256(N); per wave 16x64 (4 accumulators).
// Software-pipelined k-loop with the last step peeled: the steady-state body is
// 10 unconditional b128 loads (next step) + 4 wmmas (current step), no branches/movs.
template <bool RELU, bool ABF, bool OUTBF>
__global__ __launch_bounds__(256) void gemm_wmma(
    const void* __restrict__ Avoid, const int* __restrict__ rowIdx,
    const __bf16* __restrict__ Wt, const float* __restrict__ bias,
    void* __restrict__ outv, int M, int N, int K)
{
    const int wave = threadIdx.x >> 5;
    const int lane = threadIdx.x & 31;
    const int mw = wave & 1;        // 0..1
    const int nw = wave >> 1;       // 0..3
    const int mbase = blockIdx.x * 32 + mw * 16;
    const int nbase = blockIdx.y * 256 + nw * 64;
    if (mbase >= M || nbase >= N) return;

    const int hi = lane >> 4;
    const int ml = lane & 15;

    const int m = mbase + ml;
    const long arow = rowIdx ? (long)rowIdx[m] : (long)m;
    const float*  __restrict__ Af = (const float*)Avoid + arow * (long)K;
    const __bf16* __restrict__ Ab = (const __bf16*)Avoid + arow * (long)K;

    v8f acc[4];
    acc[0] = v8f{}; acc[1] = v8f{}; acc[2] = v8f{}; acc[3] = v8f{};

    const size_t c0 = (size_t)(nbase + 0 * 16 + ml);
    const size_t c1 = (size_t)(nbase + 1 * 16 + ml);
    const size_t c2 = (size_t)(nbase + 2 * 16 + ml);
    const size_t c3 = (size_t)(nbase + 3 * 16 + ml);

    // prologue fragments (k = 0)
    v16bf af = ABF ? load_a_bf16(Ab, 0, hi) : load_a_f32(Af, 0, hi);
    v16bf b0 = load_b(Wt, c0, K, 0, hi);
    v16bf b1 = load_b(Wt, c1, K, 0, hi);
    v16bf b2 = load_b(Wt, c2, K, 0, hi);
    v16bf b3 = load_b(Wt, c3, K, 0, hi);

    // steady state: prefetch k+32 unconditionally, then consume k
    for (int k = 0; k + 32 < K; k += 32) {
        const int kn = k + 32;
        v16bf afn = ABF ? load_a_bf16(Ab, kn, hi) : load_a_f32(Af, kn, hi);
        v16bf b0n = load_b(Wt, c0, K, kn, hi);
        v16bf b1n = load_b(Wt, c1, K, kn, hi);
        v16bf b2n = load_b(Wt, c2, K, kn, hi);
        v16bf b3n = load_b(Wt, c3, K, kn, hi);
        acc[0] = __builtin_amdgcn_wmma_f32_16x16x32_bf16(false, af, false, b0, (short)0, acc[0], false, false);
        acc[1] = __builtin_amdgcn_wmma_f32_16x16x32_bf16(false, af, false, b1, (short)0, acc[1], false, false);
        acc[2] = __builtin_amdgcn_wmma_f32_16x16x32_bf16(false, af, false, b2, (short)0, acc[2], false, false);
        acc[3] = __builtin_amdgcn_wmma_f32_16x16x32_bf16(false, af, false, b3, (short)0, acc[3], false, false);
        af = afn; b0 = b0n; b1 = b1n; b2 = b2n; b3 = b3n;
    }
    // peeled final step
    acc[0] = __builtin_amdgcn_wmma_f32_16x16x32_bf16(false, af, false, b0, (short)0, acc[0], false, false);
    acc[1] = __builtin_amdgcn_wmma_f32_16x16x32_bf16(false, af, false, b1, (short)0, acc[1], false, false);
    acc[2] = __builtin_amdgcn_wmma_f32_16x16x32_bf16(false, af, false, b2, (short)0, acc[2], false, false);
    acc[3] = __builtin_amdgcn_wmma_f32_16x16x32_bf16(false, af, false, b3, (short)0, acc[3], false, false);

    // epilogue: C/D layout lane=col, VGPR r -> row r (lanes 0-15) / r+8 (lanes 16-31)
#pragma unroll
    for (int nf = 0; nf < 4; ++nf) {
        const int col = nbase + nf * 16 + ml;
        const float bv = bias[col];
#pragma unroll
        for (int r = 0; r < 8; ++r) {
            const int row = mbase + r + hi * 8;
            float v = acc[nf][r] + bv;
            if (RELU) v = fmaxf(v, 0.0f);
            if (OUTBF) ((__bf16*)outv)[(size_t)row * N + col] = (__bf16)v;
            else       ((float*)outv)[(size_t)row * N + col] = v;
        }
    }
}

// ---------------- gate: sigmoid(h @ wg + bg) and Gumbel-perturbed scores ----------------
__global__ __launch_bounds__(256) void gate_kernel(
    const float* __restrict__ h2, const float* __restrict__ wg, const float* __restrict__ bg,
    float* __restrict__ gate_out, float* __restrict__ perturbed, int K)
{
    const int wave = threadIdx.x >> 5;
    const int lane = threadIdx.x & 31;
    const int row = blockIdx.x * 8 + wave;
    const float* __restrict__ hr = h2 + (size_t)row * K;
    float s = 0.f;
    for (int i = lane; i < K; i += 32) s += hr[i] * wg[i];
#pragma unroll
    for (int off = 16; off; off >>= 1) s += __shfl_xor(s, off, 32);
    if (lane == 0) {
        const float g = 1.f / (1.f + __expf(-(s + bg[0])));
        gate_out[row] = g;
        unsigned x = (unsigned)row * 747796405u + 2891336453u;
        x = ((x >> ((x >> 28u) + 4u)) ^ x) * 277803737u;
        x = (x >> 22u) ^ x;
        const float u = fmaxf((float)(x >> 8) * (1.0f / 16777216.0f), 1e-10f);
        perturbed[row] = g + 0.1f * (-__logf(-__logf(u)));
    }
}

// ---------------- per-batch top-128 (iterative argmax in LDS) + softmax weights ----------------
__global__ __launch_bounds__(256) void topk_kernel(
    const float* __restrict__ perturbed, int* __restrict__ topk_idx,
    float* __restrict__ soft_w)
{
    __shared__ float v[L_SZ];
    __shared__ float rv[256];
    __shared__ int   ri[256];
    __shared__ float selv[KSEL];
    const int b = blockIdx.x;
    const int t = threadIdx.x;

    for (int i = t; i < L_SZ; i += 256) v[i] = perturbed[(size_t)b * L_SZ + i];
    __syncthreads();

    for (int j = 0; j < KSEL; ++j) {
        float best = -1e30f; int bi = 0;
        for (int i = t; i < L_SZ; i += 256)
            if (v[i] > best) { best = v[i]; bi = i; }
        rv[t] = best; ri[t] = bi;
        __syncthreads();
        for (int off = 128; off; off >>= 1) {
            if (t < off && rv[t + off] > rv[t]) { rv[t] = rv[t + off]; ri[t] = ri[t + off]; }
            __syncthreads();
        }
        if (t == 0) {
            selv[j] = rv[0];
            topk_idx[b * KSEL + j] = ri[0];
            v[ri[0]] = -1e30f;
        }
        __syncthreads();
    }

    float e = 0.f;
    if (t < KSEL) e = __expf(selv[t] - selv[0]);
    rv[t] = e;
    __syncthreads();
    for (int off = 128; off; off >>= 1) {
        if (t < off) rv[t] += rv[t + off];
        __syncthreads();
    }
    if (t < KSEL) soft_w[b * KSEL + t] = e / rv[0];
}

// ---------------- build memory[b, s, :] (fp32 + bf16 mirror for the K-projection GEMM) ----------------
__global__ __launch_bounds__(256) void build_memory_kernel(
    const float* __restrict__ h2, const int* __restrict__ topk_idx,
    const float* __restrict__ soft_w, float* __restrict__ memory,
    __bf16* __restrict__ membf)
{
    const int s = blockIdx.x;   // 0..255
    const int b = blockIdx.y;   // 0..31
    const int t = threadIdx.x;
    float w = 0.f;
    const float* src = nullptr;
    if (s < KSEL) {
        const int idx = topk_idx[b * KSEL + s];
        w = soft_w[b * KSEL + s];
        src = h2 + ((size_t)b * L_SZ + idx) * HID;
    }
    const size_t roff = ((size_t)b * SLOTS + s) * HID;
    for (int h = t; h < HID; h += 256) {
        const float v = src ? src[h] * w : 0.f;
        memory[roff + h] = v;
        membf[roff + h] = (__bf16)v;
    }
}

// ---------------- query row indices: qidx[b] = b*L + (L-2) ----------------
__global__ void qidx_kernel(int* __restrict__ qidx)
{
    const int i = threadIdx.x;
    if (i < B_SZ) qidx[i] = i * L_SZ + (L_SZ - 2);
}

// ---------------- attention: softmax(q.k/sqrt(H)) over 256 slots, then read ----------------
__global__ __launch_bounds__(256) void attn_kernel(
    const float* __restrict__ q, const float* __restrict__ kmat,
    const float* __restrict__ memory, float* __restrict__ read_vec)
{
    __shared__ float w[SLOTS];
    __shared__ float red[256];
    const int b = blockIdx.x;
    const int t = threadIdx.x;   // slot id
    const float* __restrict__ qb = q + (size_t)b * HID;
    const float* __restrict__ kr = kmat + ((size_t)b * SLOTS + t) * HID;
    float s = 0.f;
    for (int i = 0; i < HID; ++i) s += qb[i] * kr[i];
    s *= (1.0f / 22.62741699796952f);   // 1/sqrt(512)

    red[t] = s; __syncthreads();
    for (int off = 128; off; off >>= 1) {
        if (t < off) red[t] = fmaxf(red[t], red[t + off]);
        __syncthreads();
    }
    const float mx = red[0]; __syncthreads();
    const float e = __expf(s - mx);
    red[t] = e; __syncthreads();
    for (int off = 128; off; off >>= 1) {
        if (t < off) red[t] += red[t + off];
        __syncthreads();
    }
    w[t] = e / red[0];
    __syncthreads();

    for (int h = t; h < HID; h += 256) {
        float acc = 0.f;
        for (int s2 = 0; s2 < SLOTS; ++s2)
            acc += w[s2] * memory[((size_t)b * SLOTS + s2) * HID + h];
        read_vec[(size_t)b * HID + h] = acc;
    }
}

// ---------------- host-side launch ----------------
extern "C" void kernel_launch(void* const* d_in, const int* in_sizes, int n_in,
                              void* d_out, int out_size, void* d_ws, size_t ws_size,
                              hipStream_t stream)
{
    (void)in_sizes; (void)n_in; (void)out_size; (void)ws_size;

    const int*   seq   = (const int*)  d_in[0];
    const float* embed = (const float*)d_in[1];
    const float* w1    = (const float*)d_in[2];
    const float* b1    = (const float*)d_in[3];
    const float* w2    = (const float*)d_in[4];
    const float* b2    = (const float*)d_in[5];
    const float* wg    = (const float*)d_in[6];
    const float* bg    = (const float*)d_in[7];
    const float* wq    = (const float*)d_in[8];
    const float* bq    = (const float*)d_in[9];
    const float* wk    = (const float*)d_in[10];
    const float* bk    = (const float*)d_in[11];
    const float* wo    = (const float*)d_in[12];
    const float* bo    = (const float*)d_in[13];

    // output layout: logits (32*32000) | gate_scores (32*4096) | h (32*4096*512)
    float* logits   = (float*)d_out;
    float* gate_out = logits + (size_t)B_SZ * VOCAB_N;
    float* h2       = gate_out + (size_t)B_SZ * L_SZ;

    // workspace carve-out (256B-aligned)
    char* base = (char*)d_ws;
    size_t off = 0;
    auto alloc = [&](size_t bytes) -> void* {
        void* p = base + off;
        off += (bytes + 255) & ~(size_t)255;
        return p;
    };
    __bf16*  h1bf      = (__bf16*) alloc((size_t)ROWS * HID * 2);           // 134 MB
    __bf16*  embed_bf  = (__bf16*) alloc((size_t)VOCAB_N * HID * 2);        // 32.8 MB
    __bf16*  w1t       = (__bf16*) alloc((size_t)HID * HID * 2);
    __bf16*  w2t       = (__bf16*) alloc((size_t)HID * HID * 2);
    __bf16*  wqt       = (__bf16*) alloc((size_t)HID * HID * 2);
    __bf16*  wkt       = (__bf16*) alloc((size_t)HID * HID * 2);
    __bf16*  wot       = (__bf16*) alloc((size_t)VOCAB_N * HID * 2);        // 32.8 MB
    float*   perturbed = (float*)  alloc((size_t)ROWS * 4);
    int*     topk_idx  = (int*)    alloc((size_t)B_SZ * KSEL * 4);
    float*   soft_w    = (float*)  alloc((size_t)B_SZ * KSEL * 4);
    float*   memory    = (float*)  alloc((size_t)B_SZ * SLOTS * HID * 4);   // 16.8 MB
    __bf16*  membf     = (__bf16*) alloc((size_t)B_SZ * SLOTS * HID * 2);   // 8.4 MB
    float*   kmat      = (float*)  alloc((size_t)B_SZ * SLOTS * HID * 4);   // 16.8 MB
    float*   qbuf      = (float*)  alloc((size_t)B_SZ * HID * 4);
    float*   read_vec  = (float*)  alloc((size_t)B_SZ * HID * 4);
    int*     qidx      = (int*)    alloc((size_t)B_SZ * 4);

    // 1) weights -> bf16 [N][K]; embed -> bf16 (row-major)
    {
        size_t n = (size_t)HID * HID;
        int blocks = (int)((n + 255) / 256);
        transpose_bf16_kernel<<<blocks, 256, 0, stream>>>(w1, w1t, HID, HID);
        transpose_bf16_kernel<<<blocks, 256, 0, stream>>>(w2, w2t, HID, HID);
        transpose_bf16_kernel<<<blocks, 256, 0, stream>>>(wq, wqt, HID, HID);
        transpose_bf16_kernel<<<blocks, 256, 0, stream>>>(wk, wkt, HID, HID);
        size_t no = (size_t)HID * VOCAB_N;
        transpose_bf16_kernel<<<(int)((no + 255) / 256), 256, 0, stream>>>(wo, wot, HID, VOCAB_N);
        convert_bf16_kernel<<<(int)((no + 255) / 256), 256, 0, stream>>>(embed, embed_bf, no);
    }

    // 2) h1 = relu(embed[seq] @ w1 + b1)  — fused gather GEMM, bf16 in / bf16 out
    gemm_wmma<true, true, true><<<dim3(ROWS / 32, HID / 256), 256, 0, stream>>>(
        embed_bf, seq, w1t, b1, h1bf, ROWS, HID, HID);

    // 3) h2 = relu(h1 @ w2 + b2)  — bf16 in, fp32 out straight into d_out's h slice
    gemm_wmma<true, true, false><<<dim3(ROWS / 32, HID / 256), 256, 0, stream>>>(
        h1bf, nullptr, w2t, b2, h2, ROWS, HID, HID);

    // 4) gate scores + perturbed
    gate_kernel<<<ROWS / 8, 256, 0, stream>>>(h2, wg, bg, gate_out, perturbed, HID);

    // 5) top-128 per batch + softmax weights
    topk_kernel<<<B_SZ, 256, 0, stream>>>(perturbed, topk_idx, soft_w);

    // 6) memory tensor (fp32 + bf16 mirror)
    build_memory_kernel<<<dim3(SLOTS, B_SZ), 256, 0, stream>>>(h2, topk_idx, soft_w, memory, membf);

    // 7) q = h2[:, L-2, :] @ wq + bq  (gather via qidx; fp32-A path, tiny)
    qidx_kernel<<<1, 32, 0, stream>>>(qidx);
    gemm_wmma<false, false, false><<<dim3(B_SZ / 32, HID / 256), 256, 0, stream>>>(
        h2, qidx, wqt, bq, qbuf, B_SZ, HID, HID);

    // 8) k = memory @ wk + bk   (M = 8192, bf16-A path)
    gemm_wmma<false, true, false><<<dim3((B_SZ * SLOTS) / 32, HID / 256), 256, 0, stream>>>(
        membf, nullptr, wkt, bk, kmat, B_SZ * SLOTS, HID, HID);

    // 9) attention read
    attn_kernel<<<B_SZ, 256, 0, stream>>>(qbuf, kmat, memory, read_vec);

    // 10) logits = read_vec @ wo + bo   (N = 32000 = 125 * 256, fp32-A path)
    gemm_wmma<false, false, false><<<dim3(B_SZ / 32, VOCAB_N / 256), 256, 0, stream>>>(
        read_vec, nullptr, wot, bo, logits, B_SZ, VOCAB_N, HID);
}
